// CausalAttention_25434796326939
// MI455X (gfx1250) — compile-verified, matrix operand order checked by reading, not run on hardware
//
#include <hip/hip_runtime.h>
#include <math.h>

#define DEVI __device__ __forceinline__

constexpr int   BATCH = 4;
constexpr int   SEQ   = 2048;
constexpr int   DIM   = 1024;          // d_in == d_out
constexpr float SCALE = 0.03125f;      // 1/sqrt(1024)

// LDS slab geometry: 128 rows x 32 k-elements (bf16), row padded to 40 elems
// (80 B) so lane r's ds_load_b128 hits bank set (20*r mod 64) -> conflict-free.
constexpr int SLAB_ROWS   = 128;
constexpr int SLAB_STRIDE = 40;                      // elements
constexpr int SLAB_ELEMS  = SLAB_ROWS * SLAB_STRIDE; // 5120

typedef __attribute__((ext_vector_type(16))) __bf16 v16bf;
typedef __attribute__((ext_vector_type(8)))  float  v8f;

union FragU { uint4 u[2]; v16bf v; };

#define ASYNC_WAIT(n) asm volatile("s_wait_asynccnt %0" :: "n"(n) : "memory")

DEVI unsigned short f32_to_bf16(float f) {
  union { float f; unsigned int u; } v; v.f = f;
  unsigned int r = v.u + 0x7FFFu + ((v.u >> 16) & 1u);   // round-to-nearest-even
  return (unsigned short)(r >> 16);
}

DEVI v8f zero_v8f() {
  v8f z;
#pragma unroll
  for (int i = 0; i < 8; ++i) z[i] = 0.0f;
  return z;
}

// Async (ASYNCcnt-tracked) 16-byte global -> LDS copy, no VGPR transit.
// Generic LDS pointer's low 32 bits are the LDS byte address (ISA 10.2 aperture
// rule).
DEVI void async_cp16(unsigned short* lds_dst, const unsigned short* gsrc) {
  const unsigned int laddr = (unsigned int)(unsigned long long)lds_dst;
  asm volatile("global_load_async_to_lds_b128 %0, %1, off"
               :: "v"(laddr), "v"(gsrc) : "memory");
}

// Stage one 128x32 bf16 slab (512 x 16B chunks) with 256 threads: 2 chunks each.
DEVI void stage_slab(unsigned short* sh, const unsigned short* gsrc, int ld) {
  const int t = threadIdx.x;
#pragma unroll
  for (int u = 0; u < 2; ++u) {
    const int c   = t + u * 256;
    const int row = c >> 2;
    const int seg = (c & 3) * 8;
    async_cp16(&sh[row * SLAB_STRIDE + seg], gsrc + (size_t)row * ld + seg);
  }
}

// Read a 16x32 bf16 WMMA operand fragment from an LDS slab (row-major tile at
// `sh`). ISA 16-bit A layout: lanes 0-15 rows 0-15 K=[0..7],[16..23];
// lanes 16-31 same rows K=[8..15],[24..31].
DEVI v16bf lds_frag(const unsigned short* sh) {
  const int lane = threadIdx.x & 31;
  const unsigned short* p = sh + (lane & 15) * SLAB_STRIDE + (lane >> 4) * 8;
  FragU f;
  f.u[0] = *(const uint4*)(p);
  f.u[1] = *(const uint4*)(p + 16);
  return f.v;
}

DEVI v8f wmma_bf16(v16bf a, v16bf b, v8f c) {
  return __builtin_amdgcn_wmma_f32_16x16x32_bf16(false, a, false, b,
                                                 (short)0, c, false, false);
}

// ---------------------------------------------------------------------------
// fp32 -> bf16 conversion, vectorized (float4 in, ushort4 out)
// ---------------------------------------------------------------------------
__global__ __launch_bounds__(256)
void k_cvt_bf16(const float* __restrict__ src, unsigned short* __restrict__ dst,
                int n4) {
  int i = blockIdx.x * blockDim.x + threadIdx.x;
  const int stride = gridDim.x * blockDim.x;
  for (; i < n4; i += stride) {
    const float4 f = ((const float4*)src)[i];
    ushort4 o;
    o.x = f32_to_bf16(f.x); o.y = f32_to_bf16(f.y);
    o.z = f32_to_bf16(f.z); o.w = f32_to_bf16(f.w);
    ((ushort4*)dst)[i] = o;
  }
}

// ---------------------------------------------------------------------------
// Projection: dst = A(8192x1024) * W(1024x1024)^T, bf16 out.
// Block = 128x128 tile, 8 waves in a 4x2 grid; wave tile 32x64 (8 WMMA acc).
// Double-buffered async global->LDS pipeline over the K dimension.
// transposed==1 stores Vt layout (B, DIM, SEQ).
// ---------------------------------------------------------------------------
__global__ __launch_bounds__(256)
void k_proj_gemm(const unsigned short* __restrict__ A,
                 const unsigned short* __restrict__ W,
                 unsigned short* __restrict__ dst,
                 int transposed) {
  __shared__ unsigned short shA[2][SLAB_ELEMS];
  __shared__ unsigned short shB[2][SLAB_ELEMS];

  const int m0b = (blockIdx.x >> 3) * 128;     // 64 M-blocks
  const int n0b = (blockIdx.x & 7) * 128;      // 8 N-blocks
  const int wave = threadIdx.x >> 5;
  const int wm = wave >> 1;                    // 0..3
  const int wn = wave & 1;                     // 0..1
  const int lane = threadIdx.x & 31;
  const int h = lane >> 4, c = lane & 15;

  v8f acc[2][4];
#pragma unroll
  for (int mi = 0; mi < 2; ++mi)
#pragma unroll
    for (int j = 0; j < 4; ++j) acc[mi][j] = zero_v8f();

  auto stage = [&](int kt, int buf) {
    stage_slab(shA[buf], A + (size_t)m0b * DIM + kt * 32, DIM);
    stage_slab(shB[buf], W + (size_t)n0b * DIM + kt * 32, DIM);
  };

  const int nsteps = DIM / 32;                 // 32
  stage(0, 0);
  for (int s = 0; s < nsteps; ++s) {
    const int cur = s & 1;
    if (s + 1 < nsteps) { stage(s + 1, cur ^ 1); ASYNC_WAIT(4); }
    else                { ASYNC_WAIT(0); }
    __syncthreads();
    v16bf a0 = lds_frag(&shA[cur][(wm * 32) * SLAB_STRIDE]);
    v16bf a1 = lds_frag(&shA[cur][(wm * 32 + 16) * SLAB_STRIDE]);
#pragma unroll
    for (int j = 0; j < 4; ++j) {
      v16bf b = lds_frag(&shB[cur][(wn * 64 + 16 * j) * SLAB_STRIDE]);
      acc[0][j] = wmma_bf16(a0, b, acc[0][j]);
      acc[1][j] = wmma_bf16(a1, b, acc[1][j]);
    }
    __syncthreads();
  }

#pragma unroll
  for (int mi = 0; mi < 2; ++mi) {
#pragma unroll
    for (int j = 0; j < 4; ++j) {
#pragma unroll
      for (int i = 0; i < 8; ++i) {
        const int row = m0b + wm * 32 + mi * 16 + i + 8 * h;  // token b*SEQ+t
        const int col = n0b + wn * 64 + 16 * j + c;           // feature
        const unsigned short v = f32_to_bf16(acc[mi][j][i]);
        if (!transposed) {
          dst[(size_t)row * DIM + col] = v;
        } else {
          const int b_ = row >> 11;
          const int t_ = row & (SEQ - 1);
          dst[(size_t)b_ * DIM * SEQ + (size_t)col * SEQ + t_] = v;
        }
      }
    }
  }
}

// ---------------------------------------------------------------------------
// Scores + causal softmax. Block = (batch, 128 q-rows); 8 waves, each owning
// 16 q-rows x 128 key-cols per group. Flattened (key-group, d-slab) async
// pipeline. Sweep 1: online max/sum stats; sweep 2: recompute + write
// P = softmax(scores) bf16, zero-padded to 128-col groups.
// ---------------------------------------------------------------------------
__global__ __launch_bounds__(256)
void k_scores_softmax(const unsigned short* __restrict__ Q,
                      const unsigned short* __restrict__ K,
                      unsigned short* __restrict__ P) {
  __shared__ unsigned short shA[2][SLAB_ELEMS];
  __shared__ unsigned short shB[2][SLAB_ELEMS];

  const int b   = blockIdx.x >> 4;             // SEQ/128 = 16 q-blocks/batch
  const int q0b = (blockIdx.x & 15) * 128;
  const int wave = threadIdx.x >> 5;
  const int lane = threadIdx.x & 31;
  const int h = lane >> 4, c = lane & 15;
  const int qw = q0b + wave * 16;              // wave's q rows

  const unsigned short* Qb = Q + (size_t)b * SEQ * DIM;
  const unsigned short* Kb = K + (size_t)b * SEQ * DIM;
  unsigned short* Pb       = P + (size_t)b * SEQ * SEQ;

  const int ngroups = q0b / 128 + 1;           // causal 128-wide key groups
  const int nsteps  = ngroups * (DIM / 32);    // flattened (group, d) steps

  auto stage = [&](int s, int buf) {
    const int kg = s >> 5;
    const int d  = (s & 31) * 32;
    stage_slab(shA[buf], Qb + (size_t)q0b * DIM + d, DIM);
    stage_slab(shB[buf], Kb + (size_t)(kg * 128) * DIM + d, DIM);
  };

  float m_run[8], l_run[8];
#pragma unroll
  for (int i = 0; i < 8; ++i) { m_run[i] = -1e30f; l_run[i] = 0.0f; }

  v8f acc[8];
#pragma unroll
  for (int j = 0; j < 8; ++j) acc[j] = zero_v8f();

  // ---- sweep 1: stats ----
  stage(0, 0);
  for (int s = 0; s < nsteps; ++s) {
    const int cur = s & 1;
    if (s + 1 < nsteps) { stage(s + 1, cur ^ 1); ASYNC_WAIT(4); }
    else                { ASYNC_WAIT(0); }
    __syncthreads();
    v16bf a = lds_frag(&shA[cur][(wave * 16) * SLAB_STRIDE]);
#pragma unroll
    for (int j = 0; j < 8; ++j) {
      v16bf bb = lds_frag(&shB[cur][(16 * j) * SLAB_STRIDE]);
      acc[j] = wmma_bf16(a, bb, acc[j]);
    }
    __syncthreads();

    if ((s & 31) == 31) {                      // end of one key group
      const int kg = s >> 5;
#pragma unroll
      for (int i = 0; i < 8; ++i) {
        const int row = qw + i + 8 * h;
        float sv[8];
#pragma unroll
        for (int j = 0; j < 8; ++j) {
          const int col = kg * 128 + 16 * j + c;
          sv[j] = (col <= row) ? acc[j][i] * SCALE : -1e30f;
        }
        float rm = sv[0];
#pragma unroll
        for (int j = 1; j < 8; ++j) rm = fmaxf(rm, sv[j]);
#pragma unroll
        for (int msk = 1; msk <= 8; msk <<= 1)
          rm = fmaxf(rm, __shfl_xor(rm, msk, 16));
        const float m_new = fmaxf(m_run[i], rm);
        float ps = 0.0f;
#pragma unroll
        for (int j = 0; j < 8; ++j) ps += __expf(sv[j] - m_new);
#pragma unroll
        for (int msk = 1; msk <= 8; msk <<= 1)
          ps += __shfl_xor(ps, msk, 16);
        l_run[i] = l_run[i] * __expf(m_run[i] - m_new) + ps;
        m_run[i] = m_new;
      }
#pragma unroll
      for (int j = 0; j < 8; ++j) acc[j] = zero_v8f();
    }
  }

  float rcp[8];
#pragma unroll
  for (int i = 0; i < 8; ++i) rcp[i] = 1.0f / l_run[i];

  // ---- sweep 2: recompute, normalize, store P ----
  stage(0, 0);
  for (int s = 0; s < nsteps; ++s) {
    const int cur = s & 1;
    if (s + 1 < nsteps) { stage(s + 1, cur ^ 1); ASYNC_WAIT(4); }
    else                { ASYNC_WAIT(0); }
    __syncthreads();
    v16bf a = lds_frag(&shA[cur][(wave * 16) * SLAB_STRIDE]);
#pragma unroll
    for (int j = 0; j < 8; ++j) {
      v16bf bb = lds_frag(&shB[cur][(16 * j) * SLAB_STRIDE]);
      acc[j] = wmma_bf16(a, bb, acc[j]);
    }
    __syncthreads();

    if ((s & 31) == 31) {
      const int kg = s >> 5;
#pragma unroll
      for (int j = 0; j < 8; ++j) {
#pragma unroll
        for (int i = 0; i < 8; ++i) {
          const int row = qw + i + 8 * h;
          const int col = kg * 128 + 16 * j + c;
          const float p = (col <= row)
                        ? __expf(acc[j][i] * SCALE - m_run[i]) * rcp[i]
                        : 0.0f;
          Pb[(size_t)row * SEQ + col] = f32_to_bf16(p);
        }
      }
#pragma unroll
      for (int j = 0; j < 8; ++j) acc[j] = zero_v8f();
    }
  }
}

// ---------------------------------------------------------------------------
// context = P(B,SEQ,SEQ) * Vt(B,DIM,SEQ)^T -> fp32 out (B,SEQ,DIM).
// Same 128x128 block pipeline as the projection; K-range limited to the
// causal region (128-col groups, matching the zero-padded P).
// ---------------------------------------------------------------------------
__global__ __launch_bounds__(256)
void k_context(const unsigned short* __restrict__ P,
               const unsigned short* __restrict__ Vt,
               float* __restrict__ out) {
  __shared__ unsigned short shA[2][SLAB_ELEMS];
  __shared__ unsigned short shB[2][SLAB_ELEMS];

  const int db = blockIdx.x & 7;               // DIM/128
  const int qb = (blockIdx.x >> 3) & 15;       // SEQ/128
  const int b  = blockIdx.x >> 7;              // batch
  const int q0b = qb * 128;
  const int d0b = db * 128;
  const int wave = threadIdx.x >> 5;
  const int wm = wave >> 1;
  const int wn = wave & 1;
  const int lane = threadIdx.x & 31;
  const int h = lane >> 4, c = lane & 15;

  const unsigned short* Pb = P  + (size_t)b * SEQ * SEQ;
  const unsigned short* Vb = Vt + (size_t)b * DIM * SEQ;
  float* outb = out + (size_t)b * SEQ * DIM;

  const int klen   = (q0b / 128 + 1) * 128;
  const int nsteps = klen / 32;

  v8f acc[2][4];
#pragma unroll
  for (int mi = 0; mi < 2; ++mi)
#pragma unroll
    for (int j = 0; j < 4; ++j) acc[mi][j] = zero_v8f();

  auto stage = [&](int s, int buf) {
    const int k = s * 32;
    stage_slab(shA[buf], Pb + (size_t)q0b * SEQ + k, SEQ);
    stage_slab(shB[buf], Vb + (size_t)d0b * SEQ + k, SEQ);
  };

  stage(0, 0);
  for (int s = 0; s < nsteps; ++s) {
    const int cur = s & 1;
    if (s + 1 < nsteps) { stage(s + 1, cur ^ 1); ASYNC_WAIT(4); }
    else                { ASYNC_WAIT(0); }
    __syncthreads();
    v16bf a0 = lds_frag(&shA[cur][(wm * 32) * SLAB_STRIDE]);
    v16bf a1 = lds_frag(&shA[cur][(wm * 32 + 16) * SLAB_STRIDE]);
#pragma unroll
    for (int j = 0; j < 4; ++j) {
      v16bf b2 = lds_frag(&shB[cur][(wn * 64 + 16 * j) * SLAB_STRIDE]);
      acc[0][j] = wmma_bf16(a0, b2, acc[0][j]);
      acc[1][j] = wmma_bf16(a1, b2, acc[1][j]);
    }
    __syncthreads();
  }

#pragma unroll
  for (int mi = 0; mi < 2; ++mi) {
#pragma unroll
    for (int j = 0; j < 4; ++j) {
#pragma unroll
      for (int i = 0; i < 8; ++i) {
        const int row = q0b + wm * 32 + mi * 16 + i + 8 * h;
        const int col = d0b + wn * 64 + 16 * j + c;
        outb[(size_t)row * DIM + col] = acc[mi][j][i];
      }
    }
  }
}

// ---------------------------------------------------------------------------
extern "C" void kernel_launch(void* const* d_in, const int* in_sizes, int n_in,
                              void* d_out, int out_size, void* d_ws, size_t ws_size,
                              hipStream_t stream) {
  (void)in_sizes; (void)n_in; (void)out_size; (void)ws_size;

  const float* x  = (const float*)d_in[0];
  const float* wq = (const float*)d_in[1];
  const float* wk = (const float*)d_in[2];
  const float* wv = (const float*)d_in[3];
  float* out = (float*)d_out;

  constexpr size_t N_X = (size_t)BATCH * SEQ * DIM;   // 8.39M
  constexpr size_t N_W = (size_t)DIM * DIM;           // 1.05M
  constexpr size_t N_P = (size_t)BATCH * SEQ * SEQ;   // 16.8M

  char* ws = (char*)d_ws;
  size_t off = 0;
  auto alloc = [&](size_t elems) -> unsigned short* {
    unsigned short* p = (unsigned short*)(ws + off);
    off = (off + elems * sizeof(unsigned short) + 255) & ~(size_t)255;
    return p;
  };
  unsigned short* xb  = alloc(N_X);
  unsigned short* wqb = alloc(N_W);
  unsigned short* wkb = alloc(N_W);
  unsigned short* wvb = alloc(N_W);
  unsigned short* Qb  = alloc(N_X);
  unsigned short* Kb  = alloc(N_X);
  unsigned short* Vtb = alloc(N_X);
  unsigned short* Pb  = alloc(N_P);

  // 1) fp32 -> bf16 (vectorized)
  k_cvt_bf16<<<2048, 256, 0, stream>>>(x,  xb,  (int)(N_X / 4));
  k_cvt_bf16<<<512,  256, 0, stream>>>(wq, wqb, (int)(N_W / 4));
  k_cvt_bf16<<<512,  256, 0, stream>>>(wk, wkb, (int)(N_W / 4));
  k_cvt_bf16<<<512,  256, 0, stream>>>(wv, wvb, (int)(N_W / 4));

  // 2) QKV projections (512 blocks each; 128x128 tiles)
  k_proj_gemm<<<512, 256, 0, stream>>>(xb, wqb, Qb,  0);
  k_proj_gemm<<<512, 256, 0, stream>>>(xb, wkb, Kb,  0);
  k_proj_gemm<<<512, 256, 0, stream>>>(xb, wvb, Vtb, 1);   // V stored transposed

  // 3) causal softmax -> P (64 blocks)
  k_scores_softmax<<<64, 256, 0, stream>>>(Qb, Kb, Pb);

  // 4) context = P @ V (512 blocks)
  k_context<<<512, 256, 0, stream>>>(Pb, Vtb, out);
}
